// LinearAttentionC_41979010351761
// MI455X (gfx1250) — matrix operand order repaired and checked by reading, not compiled, not run on hardware
//
#include <hip/hip_runtime.h>

typedef unsigned int   u32;
typedef unsigned short u16;
typedef __attribute__((ext_vector_type(16))) __bf16 v16bf;
typedef __attribute__((ext_vector_type(8)))  float  v8f;
typedef __attribute__((ext_vector_type(4)))  u32    u32x4;
typedef __attribute__((ext_vector_type(2)))  u32    u32x2;
typedef __attribute__((ext_vector_type(4)))  float  f32x4;
typedef __attribute__((ext_vector_type(8)))  u16    u16x8;

#define T_LEN 8192
#define D_IN  256
#define H_DIM 512
#define GN_N  65536.0f   // 8 channels * 8192 t per group

// ---------- optional CDNA5 async global->LDS copy -------------------
#if defined(__has_builtin)
#  if __has_builtin(__builtin_amdgcn_global_load_async_to_lds_b128) && \
      __has_builtin(__builtin_amdgcn_s_wait_asynccnt)
#    define HAVE_ASYNC_LDS 1
#  endif
#endif
#ifndef HAVE_ASYNC_LDS
#  define HAVE_ASYNC_LDS 0
#endif
#if HAVE_ASYNC_LDS
// param0: 'int __vector(4) __device__ *'  (AS1 global source)
// param1: LDS destination (AS3)
typedef int v4i_vs __attribute__((__vector_size__(16)));
typedef __attribute__((address_space(1))) v4i_vs* gptr128_t;
typedef __attribute__((address_space(3))) v4i_vs* lptr128_t;
#endif

// ---------- bf16 helpers (bit-level, round-to-nearest-even) ----------
__device__ __forceinline__ u16 f2bfbits(float f) {
  u32 u = __builtin_bit_cast(u32, f);
  u32 r = u + 0x7FFFu + ((u >> 16) & 1u);
  return (u16)(r >> 16);
}
__device__ __forceinline__ float bf2f(u16 h) {
  return __builtin_bit_cast(float, ((u32)h) << 16);
}
__device__ __forceinline__ u32 pack2bf(float lo, float hi) {
  return (u32)f2bfbits(lo) | ((u32)f2bfbits(hi) << 16);
}

// ---------- WMMA fragment plumbing (wave32, 16x16x32 bf16) ----------
union Frag { v16bf v; u32x4 q[2]; };

// sA layout: row-major [rows][32 bf16], K contiguous. A[m][k]:
// lane L: m=L&15, half=L>>4. regs 0-3: k = half*8 + {0..7};  regs 4-7: k = 16 + half*8 + {0..7}
__device__ __forceinline__ void load_frag_A(Frag& f, const u16* sA, int row, int half) {
  const u16* p = sA + row * 32;
  f.q[0] = *(const u32x4*)(p + half * 8);
  f.q[1] = *(const u32x4*)(p + 16 + half * 8);
}
// sB layout: row-major [n][32 bf16] (K contiguous per output column).
// B[k][n]: lane L: n=L&15, half=L>>4. regs 0-3: k = half*8 + {0..7};  regs 4-7: k = 16 + half*8 + {0..7}
__device__ __forceinline__ void load_frag_B(Frag& f, const u16* sB, int n, int half) {
  const u16* p = sB + n * 32 + half * 8;
  f.q[0] = *(const u32x4*)(p);
  f.q[1] = *(const u32x4*)(p + 16);
}
__device__ __forceinline__ v8f wmma_bf16(const Frag& a, const Frag& b, v8f c) {
  return __builtin_amdgcn_wmma_f32_16x16x32_bf16(false, a.v, false, b.v,
                                                 (short)0, c, false, false);
}

// stage a 64x32 f32 weight tile -> sA bf16 [m][k]; 2 b128 loads / thread
__device__ __forceinline__ void stage_A_f32(u32* sAu, const float* __restrict__ W,
                                            int ldw, int rowBase, int k0, int tid) {
  #pragma unroll
  for (int i = 0; i < 2; ++i) {
    int u = tid + 256 * i;                 // 512 units of 4 floats
    int k4 = u & 7, m = u >> 3;
    f32x4 w = *(const f32x4*)&W[(size_t)(rowBase + m) * ldw + k0 + k4 * 4];
    u32x2 p; p.x = pack2bf(w.x, w.y); p.y = pack2bf(w.z, w.w);
    *(u32x2*)&sAu[m * 16 + k4 * 2] = p;
  }
}
// stage a 32x128 f32 activation tile (transposed) -> sB bf16 [n][k]
__device__ __forceinline__ void stage_B_f32(u32* sBu, const float* __restrict__ X,
                                            int k0, int t0, int tid, bool pf) {
  #pragma unroll
  for (int i = 0; i < 2; ++i) {
    int u = tid + 256 * i;                 // 512 units: (kp 0..15, n4 0..31)
    int n4 = u & 31, kp = u >> 5;
    const float* s0 = &X[(size_t)(k0 + 2 * kp) * T_LEN + t0 + n4 * 4];
    f32x4 a = *(const f32x4*)s0;
    f32x4 c = *(const f32x4*)(s0 + T_LEN);
    #pragma unroll
    for (int j = 0; j < 4; ++j)
      sBu[(n4 * 4 + j) * 16 + kp] = pack2bf(a[j], c[j]);
    if (pf) __builtin_prefetch(s0 + (size_t)32 * T_LEN, 0, 0);
  }
}

// =====================================================================
// K1: q = softmax_head(Wq @ x) * scale   -> q_s (bf16)
// =====================================================================
__global__ __launch_bounds__(256) void qproj_kernel(
    const float* __restrict__ x, const float* __restrict__ Wq,
    u16* __restrict__ q_s) {
  const int ttile = blockIdx.x, head = blockIdx.y, b = blockIdx.z;
  const int tid = threadIdx.x, lane = tid & 31, w = tid >> 5;
  const int half = lane >> 4, lrow = lane & 15;
  const int mi = w & 3, ni = w >> 2;

  __shared__ __align__(16) u32 sAu[64 * 16];
  __shared__ __align__(16) u32 sBu[128 * 16];
  __shared__ float sC[64 * 128];
  __shared__ float sInv[128];

  const int t0 = ttile * 128;
  const int rowBase = head * 64;
  const float* xb = x + (size_t)b * D_IN * T_LEN;

  v8f acc[4] = {};
  for (int k0 = 0; k0 < 256; k0 += 32) {
    stage_A_f32(sAu, Wq, 256, rowBase, k0, tid);
    stage_B_f32(sBu, xb, k0, t0, tid, k0 + 32 < 256);
    __syncthreads();
    Frag a; load_frag_A(a, (const u16*)sAu, mi * 16 + lrow, half);
    #pragma unroll
    for (int j = 0; j < 4; ++j) {
      Frag bb; load_frag_B(bb, (const u16*)sBu, ni * 64 + j * 16 + lrow, half);
      acc[j] = wmma_bf16(a, bb, acc[j]);
    }
    __syncthreads();
  }
  // park accumulators for the cross-wave (64-row) softmax
  #pragma unroll
  for (int j = 0; j < 4; ++j)
    #pragma unroll
    for (int r = 0; r < 8; ++r)
      sC[(mi * 16 + r + 8 * half) * 128 + ni * 64 + j * 16 + lrow] = acc[j][r];
  __syncthreads();
  if (tid < 128) {                             // one column per thread
    const int n = tid;
    float mx = -1e30f;
    for (int m = 0; m < 64; ++m) mx = fmaxf(mx, sC[m * 128 + n]);
    float s = 0.f;
    for (int m = 0; m < 64; ++m) { float e = __expf(sC[m * 128 + n] - mx); sC[m * 128 + n] = e; s += e; }
    sInv[n] = 0.125f / s;                      // * DIM_HEAD^-0.5
  }
  __syncthreads();
  // coalesced packed-bf16 stores: 64 rows x 64 col-pairs
  #pragma unroll
  for (int i = 0; i < 16; ++i) {
    int idx = tid + 256 * i;
    int m = idx >> 6, c2 = (idx & 63) * 2;
    u32 p = pack2bf(sC[m * 128 + c2] * sInv[c2], sC[m * 128 + c2 + 1] * sInv[c2 + 1]);
    *(u32*)&q_s[((size_t)b * H_DIM + rowBase + m) * T_LEN + t0 + c2] = p;
  }
}

// =====================================================================
// K2: kv = Wkv @ cs -> k (bf16), v (bf16). 64x128 tile, K=256.
// =====================================================================
__global__ __launch_bounds__(256) void kvproj_kernel(
    const float* __restrict__ cs, const float* __restrict__ Wkv,
    u16* __restrict__ kq, u16* __restrict__ vq) {
  const int ttile = blockIdx.x, rtile = blockIdx.y, b = blockIdx.z;
  const int tid = threadIdx.x, lane = tid & 31, w = tid >> 5;
  const int half = lane >> 4, lrow = lane & 15;
  const int mi = w & 3, ni = w >> 2;

  __shared__ __align__(16) u32 sAu[64 * 16];
  __shared__ __align__(16) u32 sBu[128 * 16];
  __shared__ float sC[64 * 128];

  const int t0 = ttile * 128;
  const int rowBase = rtile * 64;
  const float* cb = cs + (size_t)b * D_IN * T_LEN;

  v8f acc[4] = {};
  for (int k0 = 0; k0 < 256; k0 += 32) {
    stage_A_f32(sAu, Wkv, 256, rowBase, k0, tid);
    stage_B_f32(sBu, cb, k0, t0, tid, k0 + 32 < 256);
    __syncthreads();
    Frag a; load_frag_A(a, (const u16*)sAu, mi * 16 + lrow, half);
    #pragma unroll
    for (int j = 0; j < 4; ++j) {
      Frag bb; load_frag_B(bb, (const u16*)sBu, ni * 64 + j * 16 + lrow, half);
      acc[j] = wmma_bf16(a, bb, acc[j]);
    }
    __syncthreads();
  }
  #pragma unroll
  for (int j = 0; j < 4; ++j)
    #pragma unroll
    for (int r = 0; r < 8; ++r)
      sC[(mi * 16 + r + 8 * half) * 128 + ni * 64 + j * 16 + lrow] = acc[j][r];
  __syncthreads();
  const bool isK = rowBase < 512;              // 64-row tiles never straddle 512
  u16* dst = isK ? kq : vq;
  const int rowOut = isK ? rowBase : rowBase - 512;
  #pragma unroll
  for (int i = 0; i < 16; ++i) {
    int idx = tid + 256 * i;
    int m = idx >> 6, c2 = (idx & 63) * 2;
    u32 p = pack2bf(sC[m * 128 + c2], sC[m * 128 + c2 + 1]);
    *(u32*)&dst[((size_t)b * H_DIM + rowOut + m) * T_LEN + t0 + c2] = p;
  }
}

// =====================================================================
// K3: per-row max of k over T (softmax-over-time stability)
// =====================================================================
__global__ __launch_bounds__(256) void kmax_kernel(
    const u16* __restrict__ kq, float* __restrict__ kmax) {
  const int row = blockIdx.x, tid = threadIdx.x;
  float m = -1e30f;
  for (int t = tid * 8; t < T_LEN; t += 2048) {
    u16x8 v = *(const u16x8*)&kq[(size_t)row * T_LEN + t];
    #pragma unroll
    for (int j = 0; j < 8; ++j) m = fmaxf(m, bf2f(v[j]));
  }
  __shared__ float red[256];
  red[tid] = m; __syncthreads();
  for (int s = 128; s > 0; s >>= 1) {
    if (tid < s) red[tid] = fmaxf(red[tid], red[tid + s]);
    __syncthreads();
  }
  if (tid == 0) kmax[row] = red[0];
}

// =====================================================================
// K4: ctx[d,e] += sum_t exp(k[d,t]-max_d) * v[e,t] ; Z[d] += sum_t exp
// =====================================================================
__global__ __launch_bounds__(256) void ctx_kernel(
    const u16* __restrict__ kq, const u16* __restrict__ vq,
    const float* __restrict__ kmax, float* __restrict__ ctx,
    float* __restrict__ Zg) {
  const int chunk = blockIdx.x, bh = blockIdx.y;
  const int b = bh >> 3, h = bh & 7;
  const int tid = threadIdx.x, lane = tid & 31, w = tid >> 5;
  const int half = lane >> 4, lrow = lane & 15;
  const int mi = w & 3, nb = (w >> 2) * 2;

  __shared__ __align__(16) u32 sEu[64 * 16];
  __shared__ __align__(16) u32 sVu[64 * 16];
  __shared__ float sZ[64];
  if (tid < 64) sZ[tid] = 0.f;
  __syncthreads();

  const int rowB = b * H_DIM + h * 64;
  const int d = tid >> 2;                       // staging row per thread
  const int tt0 = (tid & 3) * 8;
  const float km = kmax[rowB + d];
  const int t0c = chunk * 1024;

  v8f acc[2] = {};
  for (int it = 0; it < 32; ++it) {
    const int t0 = t0c + it * 32;
    const size_t off = (size_t)(rowB + d) * T_LEN + t0 + tt0;
    u16x8 kk = *(const u16x8*)&kq[off];
    u16x8 vv = *(const u16x8*)&vq[off];
    float zp = 0.f;
    u32x4 ep;
    #pragma unroll
    for (int j = 0; j < 4; ++j) {
      float e0 = __expf(bf2f(kk[2 * j]) - km);
      float e1 = __expf(bf2f(kk[2 * j + 1]) - km);
      zp += e0 + e1;
      ep[j] = pack2bf(e0, e1);
    }
    *(u32x4*)&sEu[d * 16 + tt0 / 2] = ep;
    *(u32x4*)&sVu[d * 16 + tt0 / 2] = __builtin_bit_cast(u32x4, vv);
    atomicAdd(&sZ[d], zp);                      // ds_add_f32
    __syncthreads();
    Frag a; load_frag_A(a, (const u16*)sEu, mi * 16 + lrow, half);
    #pragma unroll
    for (int j = 0; j < 2; ++j) {
      Frag bb; load_frag_B(bb, (const u16*)sVu, (nb + j) * 16 + lrow, half);
      acc[j] = wmma_bf16(a, bb, acc[j]);
    }
    __syncthreads();
  }
  #pragma unroll
  for (int j = 0; j < 2; ++j)
    #pragma unroll
    for (int r = 0; r < 8; ++r) {
      int dd = mi * 16 + r + 8 * half;
      int ee = (nb + j) * 16 + lrow;
      atomicAdd(&ctx[((size_t)bh * 64 + dd) * 64 + ee], acc[j][r]);
    }
  __syncthreads();
  if (tid < 64) atomicAdd(&Zg[bh * 64 + tid], sZ[tid]);
}

// =====================================================================
// K5a: fold Wout into normalized context:
// W2[b][o][h*64+d] = sum_e Wout[o][h*64+e] * ctx[bh][d][e] / Z[bh][d]
// =====================================================================
__global__ __launch_bounds__(256) void w2_kernel(
    const float* __restrict__ Wout, const float* __restrict__ ctx,
    const float* __restrict__ Zg, u16* __restrict__ W2) {
  const int h = blockIdx.x, b = blockIdx.y, bh = b * 8 + h;
  const int tid = threadIdx.x;
  __shared__ __align__(16) float sC[64 * 64];
  #pragma unroll
  for (int i = 0; i < 4; ++i) {
    int vi = tid + 256 * i;                     // 1024 f32x4 vectors
    int dd = vi >> 4;                           // 4 elems share d
    f32x4 c = *(const f32x4*)&ctx[(size_t)bh * 4096 + vi * 4];
    float iz = 1.0f / Zg[bh * 64 + dd];
    c.x *= iz; c.y *= iz; c.z *= iz; c.w *= iz;
    *(f32x4*)&sC[vi * 4] = c;
  }
  __syncthreads();
  for (int it = 0; it < 64; ++it) {
    int idx = tid + 256 * it;
    int o = idx >> 6, dd = idx & 63;
    const f32x4* wp = (const f32x4*)&Wout[(size_t)o * H_DIM + h * 64];
    const f32x4* cp = (const f32x4*)&sC[dd * 64];
    float s = 0.f;
    #pragma unroll
    for (int e4 = 0; e4 < 16; ++e4) {
      f32x4 wv = wp[e4], cv = cp[e4];
      s += wv.x * cv.x + wv.y * cv.y + wv.z * cv.z + wv.w * cv.w;
    }
    W2[((size_t)b * 256 + o) * H_DIM + h * 64 + dd] = f2bfbits(s);
  }
}

// =====================================================================
// K5b: y = W2 @ q_s + bout  (M=256,N=8192,K=512), GN partial stats fused
// =====================================================================
__global__ __launch_bounds__(256) void ygemm_kernel(
    const u16* __restrict__ W2, const u16* __restrict__ q_s,
    const float* __restrict__ bout, float* __restrict__ y,
    float* __restrict__ stats) {
  const int ttile = blockIdx.x, rtile = blockIdx.y, b = blockIdx.z;
  const int tid = threadIdx.x, lane = tid & 31, w = tid >> 5;
  const int half = lane >> 4, lrow = lane & 15;
  const int mi = w & 3, ni = w >> 2;

  __shared__ __align__(16) u32 sAu[64 * 16];
  __shared__ __align__(16) u32 sBu[128 * 16];
  __shared__ float sGs[8], sGss[8];
  if (tid < 8) { sGs[tid] = 0.f; sGss[tid] = 0.f; }

  const u16* Ab = W2 + (size_t)b * 256 * H_DIM;
  const u16* Bb = q_s + (size_t)b * H_DIM * T_LEN;
  const int rowBase = rtile * 64, t0 = ttile * 128;

  v8f acc[4] = {};
  for (int k0 = 0; k0 < 512; k0 += 32) {
    {   // stage A: straight bf16 copy, one 16B chunk per thread
      int m = tid >> 2, part = tid & 3;
      const u16* src = &Ab[(size_t)(rowBase + m) * H_DIM + k0 + part * 8];
      u16* dstp = (u16*)sAu + m * 32 + part * 8;
#if HAVE_ASYNC_LDS
      __builtin_amdgcn_global_load_async_to_lds_b128((gptr128_t)src, (lptr128_t)dstp, 0, 0);
#else
      *(u32x4*)dstp = *(const u32x4*)src;
#endif
    }
    {   // stage B: bf16 transpose, (kp 0..15, n8 0..15), one unit per thread
      int n8 = tid & 15, kp = tid >> 4;
      const u16* s0 = &Bb[(size_t)(k0 + 2 * kp) * T_LEN + t0 + n8 * 8];
      u16x8 a = *(const u16x8*)s0;
      u16x8 c = *(const u16x8*)(s0 + T_LEN);
      #pragma unroll
      for (int j = 0; j < 8; ++j)
        sBu[(n8 * 8 + j) * 16 + kp] = (u32)a[j] | ((u32)c[j] << 16);
      if (k0 + 32 < 512)
        __builtin_prefetch(s0 + (size_t)32 * T_LEN, 0, 0);
    }
#if HAVE_ASYNC_LDS
    __builtin_amdgcn_s_wait_asynccnt(0);
#endif
    __syncthreads();
    Frag a; load_frag_A(a, (const u16*)sAu, mi * 16 + lrow, half);
    #pragma unroll
    for (int j = 0; j < 4; ++j) {
      Frag bb; load_frag_B(bb, (const u16*)sBu, ni * 64 + j * 16 + lrow, half);
      acc[j] = wmma_bf16(a, bb, acc[j]);
    }
    __syncthreads();
  }
  // epilogue: bias, store raw y, accumulate group stats
  const int gl = mi * 2 + half;                 // all this lane's rows share one group
  float ls = 0.f, lss = 0.f;
  #pragma unroll
  for (int j = 0; j < 4; ++j)
    #pragma unroll
    for (int r = 0; r < 8; ++r) {
      int m = mi * 16 + r + 8 * half;
      int row = rowBase + m;
      int col = t0 + ni * 64 + j * 16 + lrow;
      float vv = acc[j][r] + bout[row];
      y[((size_t)b * 256 + row) * T_LEN + col] = vv;
      ls += vv; lss += vv * vv;
    }
  atomicAdd(&sGs[gl], ls);
  atomicAdd(&sGss[gl], lss);
  __syncthreads();
  if (tid < 8) {
    int g = (rowBase >> 3) + tid;
    atomicAdd(&stats[(b * 32 + g) * 2 + 0], sGs[tid]);
    atomicAdd(&stats[(b * 32 + g) * 2 + 1], sGss[tid]);
  }
}

// =====================================================================
// K6: GroupNorm finalize, in place on d_out (vectorized b128)
// =====================================================================
__global__ __launch_bounds__(256) void gn_kernel(
    float* __restrict__ y, const float* __restrict__ stats,
    const float* __restrict__ gamma, const float* __restrict__ beta) {
  const size_t i4 = (size_t)blockIdx.x * 256 + threadIdx.x;
  const size_t idx = i4 * 4;
  const int o = (int)((idx >> 13) & 255);
  const int b = (int)(idx >> 21);
  const int g = o >> 3;
  float s1 = stats[(b * 32 + g) * 2 + 0];
  float s2 = stats[(b * 32 + g) * 2 + 1];
  float mean = s1 * (1.0f / GN_N);
  float var = s2 * (1.0f / GN_N) - mean * mean;
  float rs = rsqrtf(var + 1e-5f) * gamma[o];
  float bt = beta[o] - mean * rs;
  f32x4 v = *(f32x4*)&y[idx];
  v.x = v.x * rs + bt; v.y = v.y * rs + bt;
  v.z = v.z * rs + bt; v.w = v.w * rs + bt;
  *(f32x4*)&y[idx] = v;
}

// =====================================================================
extern "C" void kernel_launch(void* const* d_in, const int* in_sizes, int n_in,
                              void* d_out, int out_size, void* d_ws, size_t ws_size,
                              hipStream_t stream) {
  const float* x    = (const float*)d_in[0];
  const float* cs   = (const float*)d_in[1];
  const float* Wq   = (const float*)d_in[2];
  const float* Wkv  = (const float*)d_in[3];
  const float* Wout = (const float*)d_in[4];
  const float* bout = (const float*)d_in[5];
  const float* gng  = (const float*)d_in[6];
  const float* gnb  = (const float*)d_in[7];
  float* y = (float*)d_out;

  char* ws = (char*)d_ws;
  size_t off = 0;
  auto alloc = [&](size_t bytes) -> void* {
    void* p = ws + off; off += (bytes + 255) & ~(size_t)255; return p;
  };
  const size_t NE = (size_t)8 * H_DIM * T_LEN;          // 33.5M elems
  u16*   q_s   = (u16*)alloc(NE * 2);                    // 64 MB
  u16*   kq    = (u16*)alloc(NE * 2);                    // 64 MB
  u16*   vq    = (u16*)alloc(NE * 2);                    // 64 MB
  float* kmaxb = (float*)alloc(4096 * 4);
  float* ctx   = (float*)alloc((size_t)64 * 64 * 64 * 4);
  float* Zg    = (float*)alloc(4096 * 4);
  float* stats = (float*)alloc(8 * 32 * 2 * 4);
  u16*   W2    = (u16*)alloc((size_t)8 * 256 * H_DIM * 2);

  (void)hipMemsetAsync(ctx, 0, (size_t)64 * 64 * 64 * 4, stream);
  (void)hipMemsetAsync(Zg, 0, 4096 * 4, stream);
  (void)hipMemsetAsync(stats, 0, 8 * 32 * 2 * 4, stream);

  qproj_kernel <<<dim3(64, 8, 8),  256, 0, stream>>>(x, Wq, q_s);
  kvproj_kernel<<<dim3(64, 16, 8), 256, 0, stream>>>(cs, Wkv, kq, vq);
  kmax_kernel  <<<4096,            256, 0, stream>>>(kq, kmaxb);
  ctx_kernel   <<<dim3(8, 64),     256, 0, stream>>>(kq, vq, kmaxb, ctx, Zg);
  w2_kernel    <<<dim3(8, 8),      256, 0, stream>>>(Wout, ctx, Zg, W2);
  ygemm_kernel <<<dim3(64, 4, 8),  256, 0, stream>>>(W2, q_s, bout, y, stats);
  gn_kernel    <<<16384,           256, 0, stream>>>(y, stats, gng, gnb);
}
